// ScaledDotProductAttention_14809047237154
// MI455X (gfx1250) — compile-verified, matrix operand order checked by reading, not run on hardware
//
#include <hip/hip_runtime.h>
#include <hip/hip_bf16.h>

// Problem constants (reference: B=2, H=16, S=2048, D=64)
#define BB 2
#define HH 16
#define SS 2048
#define DD 64

typedef __attribute__((ext_vector_type(16))) _Float16 v16h;
typedef __attribute__((ext_vector_type(8)))  float    v8f;

// LDS probs row stride in halfs: 2048 + 8 pad so consecutive rows land on
// shifted banks (1028 dwords per row; 1028 % 64 = 4 -> row+8 shifts 32 banks).
#define PSTRIDE (SS + 8)

__global__ __launch_bounds__(256) void attn_fused_kernel(
    const float* __restrict__ Q,
    const float* __restrict__ K,
    const float* __restrict__ V,
    const unsigned char* __restrict__ Mask,   // [B,1,S,S] bool (1 byte)
    float* __restrict__ ctx_out,              // [B,H,S,D]
    float* __restrict__ attn_out)             // [B,H,S,S]
{
    __shared__ _Float16 probs[32 * PSTRIDE];  // ~132 KB (WGP LDS = 320 KB)
    __shared__ float    rowsum[32];
    __shared__ float    rowinv[32];

    const int tid   = threadIdx.x;
    const int lane  = tid & 31;
    const int wave  = tid >> 5;              // 8 waves (wave32)
    const int l16   = lane & 15;
    const int h16   = lane >> 4;             // 0 or 1 (half-wave)

    const int nQT   = SS / 32;
    const int qt    = blockIdx.x % nQT;
    const int bh    = blockIdx.x / nQT;      // b*H + h
    const int b     = bh / HH;
    const int qbase = qt * 32;

    const float* Qp = Q + (size_t)bh * SS * DD;
    const float* Kp = K + (size_t)bh * SS * DD;
    const float* Vp = V + (size_t)bh * SS * DD;
    const unsigned char* Mp = Mask + (size_t)b * SS * SS;

    if (tid < 32) rowsum[tid] = 0.0f;
    __syncthreads();

    const int rowTile = wave & 1;            // which 16-row half of the 32-row qtile

    // ---- Load Q A-fragments (16x32 f16 each; 2 k-steps cover D=64) ----
    // ISA 16-bit A layout: lane holds row (l16); element e -> k = h16*8 + (e<8 ? e : e+8)
    const int   arow = rowTile * 16 + l16;
    const float* qrp = Qp + (size_t)(qbase + arow) * DD;
    v16h qa0, qa1;
    #pragma unroll
    for (int e = 0; e < 16; ++e) {
        const int ko = h16 * 8 + (e < 8 ? e : e + 8);
        qa0[e] = (_Float16)qrp[ko];
        qa1[e] = (_Float16)qrp[32 + ko];
    }

    // ---- Phase 1: scores -> exp -> LDS probs (unnormalized), row sums ----
    float rs[8];
    #pragma unroll
    for (int i = 0; i < 8; ++i) rs[i] = 0.0f;

    for (int ct = (wave >> 1); ct < SS / 16; ct += 4) {
        const int colbase = ct * 16;
        const float* krp = Kp + (size_t)(colbase + l16) * DD;  // key row for B-frag
        __builtin_prefetch(krp + 4 * 16 * DD, 0, 0);           // next K tile -> global_prefetch_b8

        v16h kb0, kb1;
        #pragma unroll
        for (int e = 0; e < 16; ++e) {
            const int ko = h16 * 8 + (e < 8 ? e : e + 8);
            kb0[e] = (_Float16)krp[ko];
            kb1[e] = (_Float16)krp[32 + ko];
        }

        v8f c = {};
        c = __builtin_amdgcn_wmma_f32_16x16x32_f16(false, qa0, false, kb0, (short)0, c, false, false);
        c = __builtin_amdgcn_wmma_f32_16x16x32_f16(false, qa1, false, kb1, (short)0, c, false, false);

        #pragma unroll
        for (int i = 0; i < 8; ++i) {
            const int mloc = rowTile * 16 + h16 * 8 + i;       // local row 0..31
            const int gq   = qbase + mloc;
            const float sc = c[i] * 0.125f;                    // 1/sqrt(64)
            const unsigned char mk = Mp[(size_t)gq * SS + colbase + l16];
            const float p = mk ? 0.0f : __expf(sc);            // exp(-1e9) == 0
            rs[i] += p;
            probs[mloc * PSTRIDE + colbase + l16] = (_Float16)p;
        }
    }

    // Butterfly-reduce partial row sums across the 16 lanes sharing a row set.
    #pragma unroll
    for (int i = 0; i < 8; ++i) {
        float v = rs[i];
        v += __shfl_xor(v, 1, 32);
        v += __shfl_xor(v, 2, 32);
        v += __shfl_xor(v, 4, 32);
        v += __shfl_xor(v, 8, 32);
        rs[i] = v;
    }
    if (l16 == 0) {
        #pragma unroll
        for (int i = 0; i < 8; ++i)
            atomicAdd(&rowsum[rowTile * 16 + h16 * 8 + i], rs[i]);  // ds_add_f32
    }
    __syncthreads();
    if (tid < 32) rowinv[tid] = 1.0f / rowsum[tid];
    __syncthreads();

    // ---- Phase 2: write normalized attention, coalesced float4 ----
    {
        float* abase = attn_out + (size_t)bh * SS * SS + (size_t)qbase * SS;
        #pragma unroll 4
        for (int j = 0; j < 64; ++j) {
            const int f  = tid + 256 * j;   // float4 chunk index, 32*2048/4 = 16384 total
            const int m  = f >> 9;          // / (2048/4)
            const int c4 = f & 511;
            const float inv = rowinv[m];
            const _Float16* pr = &probs[m * PSTRIDE + c4 * 4];
            float4 o;
            o.x = (float)pr[0] * inv;
            o.y = (float)pr[1] * inv;
            o.z = (float)pr[2] * inv;
            o.w = (float)pr[3] * inv;
            ((float4*)abase)[(size_t)m * (SS / 4) + c4] = o;
        }
    }

    // ---- Phase 3: context = P * V (WMMA over K=2048), scale by 1/rowsum ----
    {
        const int dimbase = (wave >> 1) * 16;     // 4 dim tiles cover D=64
        v8f acc = {};
        const int m_a = rowTile * 16 + l16;       // A-frag row (local)
        for (int kt = 0; kt < SS / 32; ++kt) {
            const int kbase = kt * 32;
            v16h pa, vb;
            #pragma unroll
            for (int e = 0; e < 16; ++e) {
                const int ko = h16 * 8 + (e < 8 ? e : e + 8);
                pa[e] = probs[m_a * PSTRIDE + kbase + ko];
                vb[e] = (_Float16)Vp[(size_t)(kbase + ko) * DD + dimbase + l16];
            }
            acc = __builtin_amdgcn_wmma_f32_16x16x32_f16(false, pa, false, vb, (short)0, acc, false, false);
        }
        float* crow = ctx_out + (size_t)bh * SS * DD + (size_t)qbase * DD;
        #pragma unroll
        for (int i = 0; i < 8; ++i) {
            const int mloc = rowTile * 16 + h16 * 8 + i;
            crow[(size_t)mloc * DD + dimbase + l16] = acc[i] * rowinv[mloc];
        }
    }
}

extern "C" void kernel_launch(void* const* d_in, const int* in_sizes, int n_in,
                              void* d_out, int out_size, void* d_ws, size_t ws_size,
                              hipStream_t stream) {
    (void)in_sizes; (void)n_in; (void)out_size; (void)d_ws; (void)ws_size;
    const float* Q = (const float*)d_in[0];
    const float* K = (const float*)d_in[1];
    const float* V = (const float*)d_in[2];
    const unsigned char* Mask = (const unsigned char*)d_in[3];
    // d_in[4] = dim_key (compile-time constant 64 here)

    float* ctx  = (float*)d_out;                        // [B,H,S,D] first
    float* attn = ctx + (size_t)BB * HH * SS * DD;      // then [B,H,S,S]

    const int grid = BB * HH * (SS / 32);               // 2048 workgroups
    attn_fused_kernel<<<grid, 256, 0, stream>>>(Q, K, V, Mask, ctx, attn);
}
